// TLSTM_26989574488313
// MI455X (gfx1250) — matrix operand order repaired and checked by reading, not compiled, not run on hardware
//
#include <hip/hip_runtime.h>
#include <stdint.h>

// ---------------- problem constants ----------------
constexpr int T_STEPS = 16384;
constexpr int IN_DIM  = 512;
constexpr int H_DIM   = 1024;
constexpr int OUT_DIM = 64;
constexpr int G4H     = 4 * H_DIM;          // 4096 gate rows (i,f,g,o)

// recurrence partitioning
constexpr int NB          = 64;             // persistent workgroups (1 per WGP)
constexpr int JPW         = H_DIM / NB;     // 16 h elements per WG
constexpr int ROWS_PER_WG = 4 * JPW;        // 64 gate rows per WG

// ---------------- CDNA5 vector types ----------------
typedef __attribute__((ext_vector_type(16))) __bf16        v16bf;
typedef __attribute__((ext_vector_type(8)))  float         v8f;
typedef __attribute__((ext_vector_type(4)))  unsigned int  u32x4;
typedef __attribute__((ext_vector_type(4)))  float         f32x4;

__device__ __forceinline__ float bflo(unsigned u) { return __uint_as_float(u << 16); }
__device__ __forceinline__ float bfhi(unsigned u) { return __uint_as_float(u & 0xffff0000u); }
__device__ __forceinline__ float sigm(float x)    { return 1.f / (1.f + __expf(-x)); }
__device__ __forceinline__ float tanh_fast(float x){ return 2.f / (1.f + __expf(-2.f * x)) - 1.f; }

// ---------------- phase 0: f32 -> bf16 (RNE) ----------------
__global__ void k_f32_to_bf16(const float* __restrict__ in,
                              unsigned short* __restrict__ out, int n) {
    int i = blockIdx.x * blockDim.x + threadIdx.x;
    int stride = gridDim.x * blockDim.x;
    for (; i < n; i += stride) {
        unsigned u = __float_as_uint(in[i]);
        out[i] = (unsigned short)((u + 0x7fffu + ((u >> 16) & 1u)) >> 16);
    }
}

// ---------------- phase 0b: zero h double-buffer + sync counter ----------------
__global__ void k_init(float* __restrict__ hbuf, unsigned* __restrict__ counter) {
    int i = blockIdx.x * blockDim.x + threadIdx.x;
    if (i < 2 * H_DIM) hbuf[i] = 0.f;
    if (i == 0) *counter = 0u;
}

// ---------------- phase 1: G[T,4096] = X[T,512] @ W_ih^T via WMMA bf16 ----------------
// block = 256 threads (8 waves); block tile = 16(M) x 512(N).
// Each wave computes a 16x64 strip: 4 accumulators, so each A fetch feeds
// 4 x v_wmma_f32_16x16x32_bf16 (4x less redundant A traffic than 1 tile/wave).
// Operand packing (contiguous k order, same in A and B):
//   A (16x32, MxK): lane L -> row = L&15; elems 0..15 = K[k0 + (L>>4)*16 ..]
//   B (32x16, KxN): lane L -> col = L&15; elems 0..15 = K[k0 + (L>>4)*16 ..]
//     B[k][n] = W_ih[n][k]  ==> a lane reads 16 *contiguous* bf16 of W_ih row n.
//   D (16x16 f32):  lane L -> col = L&15; elem e -> row = (L>>4)*8 + e
__global__ void __launch_bounds__(256) k_gemm_wmma(
        const unsigned short* __restrict__ Xb,   // [T, 512]  bf16
        const unsigned short* __restrict__ Wb,   // [4096, 512] bf16
        float* __restrict__ G)                   // [T, 4096] f32
{
    const int tid  = threadIdx.x;
    const int wave = tid >> 5;
    const int lane = tid & 31;
    const int half = lane >> 4;
    const int l15  = lane & 15;

    const int n0 = blockIdx.x * 512 + wave * 64;   // gate-row strip (4 x 16)
    const int m0 = blockIdx.y * 16;                // time-step tile

    const unsigned short* xrow = Xb + (size_t)(m0 + l15) * IN_DIM;
    const unsigned short* wrow = Wb + (size_t)(n0 + l15) * IN_DIM;

    union AB { u32x4 q[2]; v16bf v; };
    AB A, B0, B1, B2, B3;
    v8f acc0 = {}, acc1 = {}, acc2 = {}, acc3 = {};

#pragma unroll 2
    for (int k0 = 0; k0 < IN_DIM; k0 += 32) {
        A.q[0] = *(const u32x4*)(xrow + k0 + half * 16);
        A.q[1] = *(const u32x4*)(xrow + k0 + half * 16 + 8);
        const unsigned short* wk = wrow + k0 + half * 16;
        B0.q[0] = *(const u32x4*)(wk);                B0.q[1] = *(const u32x4*)(wk + 8);
        B1.q[0] = *(const u32x4*)(wk + 16 * IN_DIM);  B1.q[1] = *(const u32x4*)(wk + 16 * IN_DIM + 8);
        B2.q[0] = *(const u32x4*)(wk + 32 * IN_DIM);  B2.q[1] = *(const u32x4*)(wk + 32 * IN_DIM + 8);
        B3.q[0] = *(const u32x4*)(wk + 48 * IN_DIM);  B3.q[1] = *(const u32x4*)(wk + 48 * IN_DIM + 8);
        acc0 = __builtin_amdgcn_wmma_f32_16x16x32_bf16(false, A.v, false, B0.v, (short)0, acc0, false, false);
        acc1 = __builtin_amdgcn_wmma_f32_16x16x32_bf16(false, A.v, false, B1.v, (short)0, acc1, false, false);
        acc2 = __builtin_amdgcn_wmma_f32_16x16x32_bf16(false, A.v, false, B2.v, (short)0, acc2, false, false);
        acc3 = __builtin_amdgcn_wmma_f32_16x16x32_bf16(false, A.v, false, B3.v, (short)0, acc3, false, false);
    }

    float* grow = G + (size_t)(m0 + half * 8) * G4H + l15;
#pragma unroll
    for (int e = 0; e < 8; ++e) {
        float* ge = grow + (size_t)e * G4H;
        ge[n0]      = acc0[e];
        ge[n0 + 16] = acc1[e];
        ge[n0 + 32] = acc2[e];
        ge[n0 + 48] = acc3[e];
    }
}

// ---------------- phase 2: persistent sequential recurrence ----------------
// 64 WGs x 256 threads. WG b owns h lanes j in [b*16, b*16+16), i.e. gate rows
// {g*1024 + b*16 + j : g in 0..3}. Its 64 x 1024 bf16 W_hh slice (128 KB) lives
// in LDS for all 16384 steps. Per step:
//   - broadcast h_t (4 KB) from L2 straight into LDS with
//     global_load_async_to_lds_b128 scope:SCOPE_DEV (no VGPR round trip, no
//     stale WGP-cache hits), retired with s_wait_asynccnt 0;
//   - 4 threads per gate row reduce 256-wide K quarters from LDS, quad-shuffle;
//   - 16 threads run the pointwise LSTM cell (c state in registers);
//   - device-scope h store + global atomic arrive/spin barrier.
__global__ void __launch_bounds__(256, 1) k_recurrence(
        const unsigned short* __restrict__ Whb,  // [4096,1024] bf16
        const float* __restrict__ G,             // [T,4096] f32
        const float* __restrict__ bih,
        const float* __restrict__ bhh,
        float* __restrict__ hbuf,                // [2][1024] f32 (double buffer)
        unsigned* __restrict__ counter)
{
    extern __shared__ char smem[];
    unsigned short* Wl = (unsigned short*)smem;                        // 64*1024 bf16 = 128 KB
    float* hl = (float*)(smem + (size_t)ROWS_PER_WG * H_DIM * 2);      // 1024 f32
    float* gb = hl + H_DIM;                                            // 64 f32 gate buffer

    const int tid = threadIdx.x;
    const int b   = blockIdx.x;
    const int r   = tid >> 2;                         // 0..63 : gate row within WG
    const int sub = tid & 3;                          // K quarter
    const int gr  = (r >> 4) * H_DIM + b * JPW + (r & 15);   // global gate row
    const float bias  = bih[gr] + bhh[gr];
    const int   kbase = sub * (H_DIM / 4);

    // LDS byte offset of this thread's 16B slot in hl (aperture addr[31:0] = LDS offset)
    const unsigned hl_off  = (unsigned)(uintptr_t)hl + (unsigned)tid * 16u;
    const unsigned hl_voff = (unsigned)tid * 16u;     // byte offset within hbuf half

    // ---- stage W_hh slice into LDS (16B vector copies) ----
    for (int i = tid; i < ROWS_PER_WG * H_DIM / 8; i += 256) {
        int rr  = i >> 7;                              // row in slice (128 uint4 per row)
        int off = i & 127;
        int grr = (rr >> 4) * H_DIM + b * JPW + (rr & 15);
        ((u32x4*)Wl)[i] = ((const u32x4*)(Whb + (size_t)grr * H_DIM))[off];
    }
    __syncthreads();

    float cstate = 0.f;                                // per-j cell state (tid < 16)

    for (int t = 0; t < T_STEPS; ++t) {
        // ---- async broadcast of h_t: 256 lanes x 16B = 4 KB memory -> LDS ----
        const float* hcur = hbuf + (t & 1) * H_DIM;
        asm volatile("global_load_async_to_lds_b128 %0, %1, %2 scope:SCOPE_DEV"
                     :: "v"(hl_off), "v"(hl_voff), "s"(hcur)
                     : "memory");
        asm volatile("s_wait_asynccnt 0x0" ::: "memory");
        __syncthreads();

        // ---- 256-wide partial dot from LDS ----
        float acc = 0.f;
        const unsigned short* wr = Wl + r * H_DIM + kbase;
        const float* hr = hl + kbase;
#pragma unroll 4
        for (int k = 0; k < H_DIM / 4; k += 8) {
            u32x4 w4 = *(const u32x4*)(wr + k);
            f32x4 ha = *(const f32x4*)(hr + k);
            f32x4 hb = *(const f32x4*)(hr + k + 4);
            acc += bflo(w4.x) * ha.x + bfhi(w4.x) * ha.y
                 + bflo(w4.y) * ha.z + bfhi(w4.y) * ha.w
                 + bflo(w4.z) * hb.x + bfhi(w4.z) * hb.y
                 + bflo(w4.w) * hb.z + bfhi(w4.w) * hb.w;
        }
        // quad reduce (lanes {4q..4q+3} hold the 4 K-quarters of row r)
        acc += __shfl_xor(acc, 1, 4);
        acc += __shfl_xor(acc, 2, 4);

        if (sub == 0) {
            gb[r] = G[(size_t)t * G4H + gr] + bias + acc;   // full gate pre-activation
            __builtin_prefetch(&G[(size_t)(t + 1) * G4H + gr], 0, 1); // next step's row
        }
        __syncthreads();

        // ---- pointwise LSTM cell on 16 threads ----
        if (tid < JPW) {
            float ig = sigm(gb[tid]);
            float fg = sigm(gb[16 + tid]);
            float gg = tanh_fast(gb[32 + tid]);
            float og = sigm(gb[48 + tid]);
            cstate = fg * cstate + ig * gg;
            float hn = og * tanh_fast(cstate);
            __hip_atomic_store(&hbuf[((t + 1) & 1) * H_DIM + b * JPW + tid], hn,
                               __ATOMIC_RELAXED, __HIP_MEMORY_SCOPE_AGENT);
        }
        __syncthreads();   // h stores issued, gb consumption done

        // ---- global arrive/spin barrier across the 64 persistent WGs ----
        if (tid == 0) {
            __hip_atomic_fetch_add(counter, 1u, __ATOMIC_RELEASE, __HIP_MEMORY_SCOPE_AGENT);
            const unsigned target = (unsigned)NB * (unsigned)(t + 1);
            while (__hip_atomic_load(counter, __ATOMIC_ACQUIRE, __HIP_MEMORY_SCOPE_AGENT) < target)
                __builtin_amdgcn_s_sleep(1);
        }
        __syncthreads();
    }
}

// ---------------- phase 3: out = sigmoid(h_T @ W_lin^T + b_lin) ----------------
__global__ void k_head(const float* __restrict__ h,      // hbuf[0] = h_T (T even)
                       const float* __restrict__ Wlin,   // [64,1024]
                       const float* __restrict__ blin,   // [64]
                       float* __restrict__ out)
{
    int o = threadIdx.x;
    const float* wr = Wlin + (size_t)o * H_DIM;
    float acc = blin[o];
#pragma unroll 4
    for (int k = 0; k < H_DIM; k += 4) {
        f32x4 hv = *(const f32x4*)(h + k);
        f32x4 wv = *(const f32x4*)(wr + k);
        acc += hv.x * wv.x + hv.y * wv.y + hv.z * wv.z + hv.w * wv.w;
    }
    out[o] = sigm(acc);
}

// ---------------- launcher ----------------
extern "C" void kernel_launch(void* const* d_in, const int* in_sizes, int n_in,
                              void* d_out, int out_size, void* d_ws, size_t ws_size,
                              hipStream_t stream) {
    const float* X     = (const float*)d_in[0];
    // d_in[1] Mask, d_in[2] Delta, d_in[3] dt : unused by forward
    const float* W_ih  = (const float*)d_in[4];
    const float* W_hh  = (const float*)d_in[5];
    const float* b_ih  = (const float*)d_in[6];
    const float* b_hh  = (const float*)d_in[7];
    const float* W_lin = (const float*)d_in[8];
    const float* b_lin = (const float*)d_in[9];
    float* out = (float*)d_out;

    // workspace layout (all 256B aligned); requires ~284 MB
    char* ws = (char*)d_ws;
    unsigned short* Xb   = (unsigned short*)(ws);                          // 16 MB
    unsigned short* Wihb = (unsigned short*)(ws + ((size_t)16 << 20));     //  4 MB
    unsigned short* Whhb = (unsigned short*)(ws + ((size_t)20 << 20));     //  8 MB
    float*          G    = (float*)(ws + ((size_t)28 << 20));              // 256 MB
    float*          hbuf = (float*)(ws + ((size_t)28 << 20)
                                       + (size_t)T_STEPS * G4H * 4);       //  8 KB
    unsigned*       cnt  = (unsigned*)((char*)hbuf + 2 * H_DIM * 4);

    // phase 0: precision conversion + state init
    k_f32_to_bf16<<<1024, 256, 0, stream>>>(X,    Xb,   T_STEPS * IN_DIM);
    k_f32_to_bf16<<<512,  256, 0, stream>>>(W_ih, Wihb, G4H * IN_DIM);
    k_f32_to_bf16<<<512,  256, 0, stream>>>(W_hh, Whhb, G4H * H_DIM);
    k_init<<<8, 256, 0, stream>>>(hbuf, cnt);

    // phase 1: WMMA GEMM  G = X @ W_ih^T   (block tile 16 x 512, wave tile 16 x 64)
    k_gemm_wmma<<<dim3(G4H / 512, T_STEPS / 16), 256, 0, stream>>>(Xb, Wihb, G);

    // phase 2: persistent recurrence (128 KB W slice + 4 KB h + gate buf in LDS)
    size_t smem = (size_t)ROWS_PER_WG * H_DIM * 2 + (H_DIM + 64) * sizeof(float);
    k_recurrence<<<NB, 256, smem, stream>>>(Whhb, G, b_ih, b_hh, hbuf, cnt);

    // phase 3: linear head + sigmoid
    k_head<<<1, OUT_DIM, 0, stream>>>(hbuf, W_lin, b_lin, out);
}
